// EDRN_63548336111823
// MI455X (gfx1250) — matrix operand order repaired and checked by reading, not compiled, not run on hardware
//
#include <hip/hip_runtime.h>

typedef unsigned short u16;
typedef unsigned int   u32;
typedef __attribute__((ext_vector_type(16))) __bf16 v16bf;
typedef __attribute__((ext_vector_type(8)))  __bf16 v8bf;
typedef __attribute__((ext_vector_type(8)))  float  v8f;

#define NWG   32      // persistent workgroups; each owns a 16-col panel of the 512 c-cols
#define TBL   256     // 8 waves (wave32) per WG; wave w owns batch-row tile w (16 rows)
#define TLEN  1024
#define BSZ   128

// ---------------- workspace layout (bytes) ----------------
static constexpr size_t OFF_XBF  = 0;                         // 128*1024*128 bf16
static constexpr size_t OFF_BCM  = OFF_XBF  + 33554432;       // B  col-major bf16  (128x2048)
static constexpr size_t OFF_AFG  = OFF_BCM  + 524288;         // A_fg_in cm bf16    (128x1024)
static constexpr size_t OFF_APT  = OFF_AFG  + 262144;         // A_pt cm bf16       (512x512)
static constexpr size_t OFF_ATH  = OFF_APT  + 524288;         // A_th cm bf16       (128x512)
static constexpr size_t OFF_AOT  = OFF_ATH  + 131072;         // A_ot cm bf16       (128x512)
static constexpr size_t OFF_AST  = OFF_AOT  + 131072;         // A_st cm bf16       (128x128)
static constexpr size_t OFF_CBF0 = OFF_AST  + 32768;          // c bf16 buf0 (128x512)
static constexpr size_t OFF_CBF1 = OFF_CBF0 + 131072;         // c bf16 buf1
static constexpr size_t OFF_CCBF = OFF_CBF1 + 131072;         // cc bf16     (128x512)
static constexpr size_t OFF_HBF  = OFF_CCBF + 131072;         // h bf16      (128x128)
static constexpr size_t OFF_BAR  = OFF_HBF  + 32768;          // barrier counter

// ---------------- small helpers ----------------
static __device__ __forceinline__ u16 f2bf(float f) {
    u32 u = __float_as_uint(f);
    u = (u + 0x7FFFu + ((u >> 16) & 1u)) >> 16;   // RNE
    return (u16)u;
}
static __device__ __forceinline__ float bf2f(u16 h) {
    return __uint_as_float(((u32)h) << 16);
}
static __device__ __forceinline__ float sigm(float x) {
    return 1.0f / (1.0f + __expf(-x));
}
static __device__ __forceinline__ float tanh_f(float x) {
    float e = __expf(-2.0f * fabsf(x));
    float t = (1.0f - e) / (1.0f + e);
    return copysignf(t, x);
}
static __device__ __forceinline__ v16bf pack16(v8bf lo, v8bf hi) {
    return __builtin_shufflevector(lo, hi, 0,1,2,3,4,5,6,7,8,9,10,11,12,13,14,15);
}

// WMMA: D = A(16x32 bf16) * B(32x16 bf16) + C(16x16 f32)
static __device__ __forceinline__ v8f wmma_bf(v16bf a, v16bf b, v8f c) {
    return __builtin_amdgcn_wmma_f32_16x16x32_bf16(false, a, false, b, (short)0, c, false, false);
}

// A-operand (16x32 bf16 tile) from row-major bf16 memory.
// ISA layout: lanes 0-15 -> rows, halves 0-7 = K k..k+7, halves 8-15 = K k+16..k+23;
// lanes 16-31: same rows, K offset +8 / +24.
static __device__ __forceinline__ v16bf load_a_g(const u16* base, int ld, int row0, int k0) {
    const int lane = threadIdx.x & 31;
    const int r = row0 + (lane & 15);
    const int k = k0 + ((lane >> 4) << 3);
    const __bf16* p = reinterpret_cast<const __bf16*>(base) + (size_t)r * ld + k;
    v8bf lo = *reinterpret_cast<const v8bf*>(p);
    v8bf hi = *reinterpret_cast<const v8bf*>(p + 16);
    return pack16(lo, hi);
}

// B-operand (32x16 bf16 tile) from column-major bf16 panel (LDS).
// Lanes 0-15: col n, K k0..k0+7 / k0+16..k0+23; lanes 16-31: col n, K k0+8..k0+15 / k0+24..k0+31.
static __device__ __forceinline__ v16bf load_b_s(const u16* panel, int ldk, int k0) {
    const int lane = threadIdx.x & 31;
    const int n = lane & 15;
    const int k = k0 + ((lane >> 4) << 3);
    const __bf16* p = reinterpret_cast<const __bf16*>(panel) + (size_t)n * ldk + k;
    v8bf lo = *reinterpret_cast<const v8bf*>(p);
    v8bf hi = *reinterpret_cast<const v8bf*>(p + 16);
    return pack16(lo, hi);
}

// Monotonic device-wide split barrier (counter reset to 0 by init kernel each call).
static __device__ __forceinline__ void grid_barrier(u32* bar, u32* phase) {
    __syncthreads();
    if (threadIdx.x == 0) {
        u32 target = ++(*phase) * NWG;
        __threadfence();
        __hip_atomic_fetch_add(bar, 1u, __ATOMIC_ACQ_REL, __HIP_MEMORY_SCOPE_AGENT);
        while (__hip_atomic_load(bar, __ATOMIC_ACQUIRE, __HIP_MEMORY_SCOPE_AGENT) < target) {
            __builtin_amdgcn_s_sleep(2);
        }
    }
    __syncthreads();
}

// ---------------- phase-0 kernels ----------------
__global__ void edrn_cvt_x(const float* __restrict__ x, u16* __restrict__ xb, int n) {
    for (int i = blockIdx.x * blockDim.x + threadIdx.x; i < n; i += gridDim.x * blockDim.x)
        xb[i] = f2bf(x[i]);
}
// row-major (K x N) fp32  ->  column-major bf16: wc[n*K + k] = w[k*N + n]
__global__ void edrn_cvt_wcm(const float* __restrict__ w, u16* __restrict__ wc, int K, int N) {
    int total = K * N;
    for (int i = blockIdx.x * blockDim.x + threadIdx.x; i < total; i += gridDim.x * blockDim.x) {
        int n = i / K, k = i - n * K;
        wc[i] = f2bf(w[(size_t)k * N + n]);
    }
}
__global__ void edrn_init_bar(u32* bar) { *bar = 0u; }

// ---------------- persistent recurrent kernel ----------------
__global__ __launch_bounds__(TBL, 1) void edrn_scan(
    const u16* __restrict__ xbf,   const u16* __restrict__ bcm,
    const u16* __restrict__ afgcm, const u16* __restrict__ aptcm,
    const u16* __restrict__ athcm, const u16* __restrict__ aotcm,
    const u16* __restrict__ astcm, const float* __restrict__ bias,
    u16* __restrict__ cbf0, u16* __restrict__ cbf1,
    u16* __restrict__ ccbf, u16* __restrict__ hbf,
    u32* __restrict__ bar, float* __restrict__ out)
{
    __shared__ __attribute__((aligned(16))) u16 sB[4 * 2048];  // 4 x (16 cols x K=128) of B
    __shared__ __attribute__((aligned(16))) u16 sAfg[2048];    // 16 x 128
    __shared__ __attribute__((aligned(16))) u16 sAin[2048];    // 16 x 128
    __shared__ __attribute__((aligned(16))) u16 sApt[8192];    // 16 x 512
    __shared__ __attribute__((aligned(16))) u16 sAth[2048];    // 16 x 128
    __shared__ __attribute__((aligned(16))) u16 sAot[2048];    // 16 x 128
    __shared__ __attribute__((aligned(16))) u16 sAst[2048];    // 16 x 128
    __shared__ __attribute__((aligned(16))) float sC[128 * 16]; // fp32 master of owned c cols

    const int g    = blockIdx.x;        // owns c/cc/pre-act columns 16g..16g+15
    const int tid  = threadIdx.x;
    const int wave = tid >> 5;
    const int lane = tid & 31;
    const int rt16 = wave * 16;         // batch-row tile
    const int colL = lane & 15;
    const int gcol = g * 16 + colL;
    u32 phase = 0;

    // --- load this WG's weight panels into LDS (already bf16 column-major) ---
    #pragma unroll
    for (int p = 0; p < 4; ++p) {
        const u16* src = bcm + (size_t)(p * 512 + g * 16) * 128;
        for (int i = tid; i < 2048; i += TBL) sB[p * 2048 + i] = src[i];
    }
    { const u16* s = afgcm + (size_t)(g * 16) * 128;        for (int i = tid; i < 2048; i += TBL) sAfg[i] = s[i]; }
    { const u16* s = afgcm + (size_t)(512 + g * 16) * 128;  for (int i = tid; i < 2048; i += TBL) sAin[i] = s[i]; }
    { const u16* s = aptcm + (size_t)(g * 16) * 512;        for (int i = tid; i < 8192; i += TBL) sApt[i] = s[i]; }
    { const u16* s = athcm + (size_t)(g * 16) * 128;        for (int i = tid; i < 2048; i += TBL) sAth[i] = s[i]; }
    { const u16* s = aotcm + (size_t)(g * 16) * 128;        for (int i = tid; i < 2048; i += TBL) sAot[i] = s[i]; }
    if (g < 8) { const u16* s = astcm + (size_t)(g * 16) * 128; for (int i = tid; i < 2048; i += TBL) sAst[i] = s[i]; }

    // --- zero initial state ---
    for (int i = tid; i < 2048; i += TBL) sC[i] = 0.0f;
    for (int i = tid; i < 2048; i += TBL) {
        int r = i >> 4, cl = i & 15;
        cbf0[(size_t)r * 512 + g * 16 + cl] = 0;
        cbf1[(size_t)r * 512 + g * 16 + cl] = 0;
    }
    if (g < 8)
        for (int i = tid; i < 2048; i += TBL) {
            int r = i >> 4, cl = i & 15;
            hbf[(size_t)r * 128 + g * 16 + cl] = 0;
        }
    __threadfence();
    grid_barrier(bar, &phase);

    float* out_hs = out;                          // (128, 1024, 128)
    float* out_h  = out + (size_t)128 * 1024 * 128;
    float* out_c  = out_h + 128 * 128;

    for (int t = 0; t < TLEN; ++t) {
        const u16* c_rd = (t & 1) ? cbf1 : cbf0;   // c from step t-1
        u16*       c_wr = (t & 1) ? cbf0 : cbf1;   // c of step t

        // ---- b' = x_t @ B + b : 4 owned 16-col tiles, kept in registers ----
        v8f bp[4];
        const u16* xb = xbf + (size_t)t * 128;     // row stride = T*N = 131072
        #pragma unroll
        for (int p = 0; p < 4; ++p) {
            v8f acc = {};
            #pragma unroll
            for (int k0 = 0; k0 < 128; k0 += 32)
                acc = wmma_bf(load_a_g(xb, 131072, rt16, k0),
                              load_b_s(&sB[p * 2048], 128, k0), acc);
            float bv = bias[p * 512 + gcol];
            #pragma unroll
            for (int i = 0; i < 8; ++i) acc[i] += bv;
            bp[p] = acc;
        }
        if (t + 1 < TLEN)   // pull next step's x rows toward L2/L0
            __builtin_prefetch(xbf + (size_t)rt16 * 131072 + (size_t)(t + 1) * 128, 0, 1);

        // ---- stage 1: fg/in pre-acts + pt pre-act; elementwise c update ----
        v8f afg = bp[0], ain = bp[1], pt = bp[2];
        #pragma unroll
        for (int k0 = 0; k0 < 128; k0 += 32) {
            v16bf ah = load_a_g(hbf, 128, rt16, k0);
            afg = wmma_bf(ah, load_b_s(sAfg, 128, k0), afg);
            ain = wmma_bf(ah, load_b_s(sAin, 128, k0), ain);
        }
        #pragma unroll 4
        for (int k0 = 0; k0 < 512; k0 += 32)        // c @ A_pt (K = 512)
            pt = wmma_bf(load_a_g(c_rd, 512, rt16, k0), load_b_s(sApt, 512, k0), pt);
        #pragma unroll
        for (int k0 = 0; k0 < 128; k0 += 32)        // c[:, -128:] @ A_th
            pt = wmma_bf(load_a_g(c_rd, 512, rt16, 384 + k0), load_b_s(sAth, 128, k0), pt);

        #pragma unroll
        for (int i = 0; i < 8; ++i) {
            int row = rt16 + i + ((lane >> 4) << 3);
            float cold = sC[row * 16 + colL];
            float cn = cold * sigm(afg[i]) + tanh_f(pt[i]) * sigm(ain[i]);
            sC[row * 16 + colL] = cn;
            c_wr[(size_t)row * 512 + gcol] = f2bf(cn);
        }
        grid_barrier(bar, &phase);

        // ---- stage 2: o-gate from NEW c last block; cc = tanh(c)*sig(ot) ----
        v8f ot = bp[3];
        #pragma unroll
        for (int k0 = 0; k0 < 128; k0 += 32)
            ot = wmma_bf(load_a_g(c_wr, 512, rt16, 384 + k0), load_b_s(sAot, 128, k0), ot);
        #pragma unroll
        for (int i = 0; i < 8; ++i) {
            int row = rt16 + i + ((lane >> 4) << 3);
            float cc = tanh_f(sC[row * 16 + colL]) * sigm(ot[i]);
            ccbf[(size_t)row * 512 + gcol] = f2bf(cc);
        }
        grid_barrier(bar, &phase);

        // ---- stage 3 (WGs 0..7): h = cc_last @ A_st + sum of first 3 cc blocks ----
        if (g < 8) {
            v8f hv = {};
            #pragma unroll
            for (int k0 = 0; k0 < 128; k0 += 32)
                hv = wmma_bf(load_a_g(ccbf, 512, rt16, 384 + k0), load_b_s(sAst, 128, k0), hv);
            #pragma unroll
            for (int i = 0; i < 8; ++i) {
                int row = rt16 + i + ((lane >> 4) << 3);
                size_t rb = (size_t)row * 512 + gcol;
                float s = hv[i] + bf2f(ccbf[rb]) + bf2f(ccbf[rb + 128]) + bf2f(ccbf[rb + 256]);
                hbf[(size_t)row * 128 + gcol] = f2bf(s);
                out_hs[((size_t)row * 1024 + t) * 128 + gcol] = s;
                if (t == TLEN - 1) out_h[(size_t)row * 128 + gcol] = s;
            }
        }
        grid_barrier(bar, &phase);
    }

    // final c_t (fp32 master lives in owner's LDS)
    for (int i = tid; i < 2048; i += TBL) {
        int r = i >> 4, cl = i & 15;
        out_c[(size_t)r * 512 + g * 16 + cl] = sC[i];
    }
}

// ---------------- host launcher ----------------
extern "C" void kernel_launch(void* const* d_in, const int* in_sizes, int n_in,
                              void* d_out, int out_size, void* d_ws, size_t ws_size,
                              hipStream_t stream) {
    const float* x    = (const float*)d_in[0];
    const float* A_th = (const float*)d_in[1];
    const float* A_ot = (const float*)d_in[2];
    const float* A_fg = (const float*)d_in[3];
    const float* A_pt = (const float*)d_in[4];
    const float* A_st = (const float*)d_in[5];
    const float* B    = (const float*)d_in[6];
    const float* bias = (const float*)d_in[7];

    char* ws = (char*)d_ws;
    u16* xbf   = (u16*)(ws + OFF_XBF);
    u16* bcm   = (u16*)(ws + OFF_BCM);
    u16* afgcm = (u16*)(ws + OFF_AFG);
    u16* aptcm = (u16*)(ws + OFF_APT);
    u16* athcm = (u16*)(ws + OFF_ATH);
    u16* aotcm = (u16*)(ws + OFF_AOT);
    u16* astcm = (u16*)(ws + OFF_AST);
    u16* cbf0  = (u16*)(ws + OFF_CBF0);
    u16* cbf1  = (u16*)(ws + OFF_CBF1);
    u16* ccbf  = (u16*)(ws + OFF_CCBF);
    u16* hbf   = (u16*)(ws + OFF_HBF);
    u32* bar   = (u32*)(ws + OFF_BAR);

    edrn_cvt_x  <<<512, 256, 0, stream>>>(x, xbf, BSZ * TLEN * 128);
    edrn_cvt_wcm<<<128, 256, 0, stream>>>(B,    bcm,   128, 2048);
    edrn_cvt_wcm<<< 64, 256, 0, stream>>>(A_fg, afgcm, 128, 1024);
    edrn_cvt_wcm<<<128, 256, 0, stream>>>(A_pt, aptcm, 512,  512);
    edrn_cvt_wcm<<< 32, 256, 0, stream>>>(A_th, athcm, 128,  512);
    edrn_cvt_wcm<<< 32, 256, 0, stream>>>(A_ot, aotcm, 128,  512);
    edrn_cvt_wcm<<<  8, 256, 0, stream>>>(A_st, astcm, 128,  128);
    edrn_init_bar<<<1, 1, 0, stream>>>(bar);

    edrn_scan<<<NWG, TBL, 0, stream>>>(xbf, bcm, afgcm, aptcm, athcm, aotcm, astcm, bias,
                                       cbf0, cbf1, ccbf, hbf, bar, (float*)d_out);
}
